// Seq2SeqLSTM_6399501271823
// MI455X (gfx1250) — compile-verified, matrix operand order checked by reading, not run on hardware
//
#include <hip/hip_runtime.h>

typedef __attribute__((ext_vector_type(16))) _Float16 v16h;
typedef __attribute__((ext_vector_type(8)))  _Float16 v8h;
typedef __attribute__((ext_vector_type(8)))  float    v8f;
typedef __attribute__((ext_vector_type(4)))  float    v4f;

#define HID   256
#define BATCH 2048

// ---------------------------------------------------------------------------
// Prep: zero h/c state (poisoned workspace), build WMMA-fragment-swizzled f16
// weights, fuse biases.
// ---------------------------------------------------------------------------
__global__ void zero_words(unsigned int* __restrict__ p, int n) {
    int i = blockIdx.x * blockDim.x + threadIdx.x;
    if (i < n) p[i] = 0u;
}

// Encoder weights: gates = [x|h] @ [W_ih|W_hh]^T, K = 64 + 256 = 320 (KB=10).
// Fragment layout: Wf[(((nb*KB)+kb)*32 + lane)*16 + e], n = nb*16 + (lane&15),
// k = kb*32 + (e<8 ? khalf+e : 16+khalf+(e-8)), khalf = (lane>>4)*8.
__global__ void prep_wenc(const float* __restrict__ Wih,   // [1024 x 64]
                          const float* __restrict__ Whh,   // [1024 x 256]
                          _Float16* __restrict__ Wf) {
    int idx = blockIdx.x * 256 + threadIdx.x;        // exact: 64*10*32*16
    int e = idx & 15;
    int t = idx >> 4;
    int lane = t & 31; t >>= 5;
    int kb = t % 10;
    int nb = t / 10;
    int n = nb * 16 + (lane & 15);
    int khalf = (lane >> 4) * 8;
    int k = kb * 32 + (e < 8 ? khalf + e : 16 + khalf + (e - 8));
    float w = (k < 64) ? Wih[n * 64 + k] : Whh[n * 256 + (k - 64)];
    Wf[idx] = (_Float16)w;
}

// Decoder: input == h, so fold W_ih+W_hh. K = 256 (KB=8).
__global__ void prep_wdec(const float* __restrict__ Wih,   // [1024 x 256]
                          const float* __restrict__ Whh,   // [1024 x 256]
                          _Float16* __restrict__ Wf) {
    int idx = blockIdx.x * 256 + threadIdx.x;        // exact: 64*8*32*16
    int e = idx & 15;
    int t = idx >> 4;
    int lane = t & 31; t >>= 5;
    int kb = t % 8;
    int nb = t / 8;
    int n = nb * 16 + (lane & 15);
    int khalf = (lane >> 4) * 8;
    int k = kb * 32 + (e < 8 ? khalf + e : 16 + khalf + (e - 8));
    Wf[idx] = (_Float16)(Wih[n * 256 + k] + Whh[n * 256 + k]);
}

__global__ void prep_bias(const float* __restrict__ bie, const float* __restrict__ bhe,
                          const float* __restrict__ bid, const float* __restrict__ bhd,
                          float* __restrict__ bE, float* __restrict__ bD) {
    int i = blockIdx.x * 256 + threadIdx.x;          // 2048
    if (i < 1024) bE[i] = bie[i] + bhe[i];
    else          bD[i - 1024] = bid[i - 1024] + bhd[i - 1024];
}

// ---------------------------------------------------------------------------
// One LSTM timestep. Grid (32, 16): blockIdx.x = 64-row batch tile,
// blockIdx.y = 16-unit hidden tile. 128 threads = 4 waves; wave g computes
// gate g's 64x16 strip with v_wmma_f32_16x16x32_f16, then the block fuses the
// elementwise cell update via LDS. Gates never touch global memory.
// ---------------------------------------------------------------------------
template <int KB, bool ENC>
__global__ __launch_bounds__(128) void lstm_step(
    const float*    __restrict__ xt,    // [2048 x 64] (ENC only)
    const _Float16* __restrict__ Hin,   // [2048 x 256]
    _Float16*       __restrict__ Hout,  // [2048 x 256]
    float*          __restrict__ C,     // [2048 x 256] fp32 cell state
    const _Float16* __restrict__ Wf,    // fragment-swizzled weights
    const float*    __restrict__ bias)  // [1024] = b_ih + b_hh
{
    __shared__ float ldsG[4][64 * 16];  // 16 KB: i,f,g,o tiles

    const int tid   = threadIdx.x;
    const int lane  = tid & 31;
    const int g     = tid >> 5;                 // gate index (i,f,g,o)
    const int mBase = blockIdx.x * 64;
    const int hb    = blockIdx.y;
    const int n0    = hb * 16;
    const int nb    = g * 16 + hb;              // W-row block: rows g*256+n0..+15
    const int mrow  = lane & 15;
    const int khalf = (lane >> 4) * 8;

    const v8f zero8 = {0.f, 0.f, 0.f, 0.f, 0.f, 0.f, 0.f, 0.f};
    v8f acc[4] = {zero8, zero8, zero8, zero8};

    const _Float16* __restrict__ wlane = Wf + ((size_t)(nb * KB) * 32 + lane) * 16;

#pragma unroll
    for (int kb = 0; kb < KB; ++kb) {
        // B fragment: 32 contiguous bytes per lane (pre-swizzled)
        const _Float16* wq = wlane + kb * (32 * 16);
        v8h blo = *(const v8h*)(wq);
        v8h bhi = *(const v8h*)(wq + 8);
        v16h bfrag = __builtin_shufflevector(blo, bhi,
            0,1,2,3,4,5,6,7,8,9,10,11,12,13,14,15);

#pragma unroll
        for (int mb = 0; mb < 4; ++mb) {
            const int row = mBase + mb * 16 + mrow;
            v16h afrag;
            if (ENC && kb < 2) {
                // x part (fp32 -> f16 on the fly), K 0..63
                const float* p = xt + row * 64 + kb * 32 + khalf;
                v4f f0 = *(const v4f*)(p);
                v4f f1 = *(const v4f*)(p + 4);
                v4f f2 = *(const v4f*)(p + 16);
                v4f f3 = *(const v4f*)(p + 20);
#pragma unroll
                for (int i = 0; i < 4; ++i) {
                    afrag[i]      = (_Float16)f0[i];
                    afrag[4 + i]  = (_Float16)f1[i];
                    afrag[8 + i]  = (_Float16)f2[i];
                    afrag[12 + i] = (_Float16)f3[i];
                }
            } else {
                const int kh = (ENC ? (kb - 2) : kb) * 32 + khalf;
                const _Float16* p = Hin + row * HID + kh;
                v8h alo = *(const v8h*)(p);
                v8h ahi = *(const v8h*)(p + 16);
                afrag = __builtin_shufflevector(alo, ahi,
                    0,1,2,3,4,5,6,7,8,9,10,11,12,13,14,15);
            }
            acc[mb] = __builtin_amdgcn_wmma_f32_16x16x32_f16(
                false, afrag, false, bfrag, (short)0, acc[mb], false, false);
        }
    }

    // C/D layout: lane L, vgpr v -> (M = v + (L>=16 ? 8 : 0), N = L&15)
    const int mhi = (lane >> 4) * 8;
#pragma unroll
    for (int mb = 0; mb < 4; ++mb)
#pragma unroll
        for (int v = 0; v < 8; ++v)
            ldsG[g][(mb * 16 + v + mhi) * 16 + (lane & 15)] = acc[mb][v];

    __syncthreads();

    // Fused elementwise LSTM cell update: 1024 elements / 128 threads.
#pragma unroll
    for (int j = 0; j < 8; ++j) {
        const int e   = j * 128 + tid;          // bank-conflict-free stride
        const int ml  = e >> 4;
        const int nl  = e & 15;
        const int row = mBase + ml;
        const int col = n0 + nl;
        const float gi = ldsG[0][e] + bias[col];
        const float gf = ldsG[1][e] + bias[256 + col];
        const float gg = ldsG[2][e] + bias[512 + col];
        const float go = ldsG[3][e] + bias[768 + col];
        const float si = 1.f / (1.f + expf(-gi));
        const float sf = 1.f / (1.f + expf(-gf));
        const float so = 1.f / (1.f + expf(-go));
        const float cN = sf * C[row * HID + col] + si * tanhf(gg);
        C[row * HID + col]    = cN;
        Hout[row * HID + col] = (_Float16)(so * tanhf(cN));
    }
}

// pred[b] = h[b,:] . W_lin + b_lin   (O = 1)
__global__ void linear_out(const _Float16* __restrict__ H,
                           const float* __restrict__ Wl,
                           const float* __restrict__ bl,
                           float* __restrict__ out) {
    int b = blockIdx.x * 256 + threadIdx.x;     // exact 2048
    const _Float16* hp = H + (size_t)b * HID;
    float s = bl[0];
#pragma unroll 8
    for (int j = 0; j < HID; ++j) s += (float)hp[j] * Wl[j];
    out[b] = s;
}

// ---------------------------------------------------------------------------
extern "C" void kernel_launch(void* const* d_in, const int* in_sizes, int n_in,
                              void* d_out, int out_size, void* d_ws, size_t ws_size,
                              hipStream_t stream) {
    (void)in_sizes; (void)n_in; (void)out_size; (void)ws_size;

    const float* x      = (const float*)d_in[0];   // [168, 2048, 64]
    const float* W_ih_e = (const float*)d_in[1];   // [1024, 64]
    const float* W_hh_e = (const float*)d_in[2];   // [1024, 256]
    const float* b_ih_e = (const float*)d_in[3];
    const float* b_hh_e = (const float*)d_in[4];
    const float* W_ih_d = (const float*)d_in[5];   // [1024, 256]
    const float* W_hh_d = (const float*)d_in[6];   // [1024, 256]
    const float* b_ih_d = (const float*)d_in[7];
    const float* b_hh_d = (const float*)d_in[8];
    const float* W_lin  = (const float*)d_in[9];   // [1, 256]
    const float* b_lin  = (const float*)d_in[10];  // [1]
    float* out = (float*)d_out;                    // [24, 2048, 1]

    // Workspace layout (~5.4 MB):
    char* ws = (char*)d_ws;
    _Float16* H0  = (_Float16*)(ws);                               // 1 MB
    _Float16* H1  = (_Float16*)(ws + (1u << 20));                  // 1 MB
    float*    C   = (float*)   (ws + (2u << 20));                  // 2 MB
    _Float16* WfE = (_Float16*)(ws + (4u << 20));                  // 640 KB
    _Float16* WfD = (_Float16*)(ws + (4u << 20) + 655360u);        // 512 KB
    float*    bE  = (float*)   (ws + (4u << 20) + 655360u + 524288u);
    float*    bD  = (float*)   (ws + (4u << 20) + 655360u + 524288u + 4096u);

    // Re-init state every launch (ws is poisoned, not restored between replays).
    zero_words<<<4096, 256, 0, stream>>>((unsigned int*)ws, 1 << 20); // H0,H1,C
    prep_wenc<<<1280, 256, 0, stream>>>(W_ih_e, W_hh_e, WfE);         // 64*10*32*16
    prep_wdec<<<1024, 256, 0, stream>>>(W_ih_d, W_hh_d, WfD);         // 64*8*32*16
    prep_bias<<<8, 256, 0, stream>>>(b_ih_e, b_hh_e, b_ih_d, b_hh_d, bE, bD);

    _Float16* Hc = H0;
    _Float16* Hn = H1;

    // Encoder: 168 sequential steps, K = 320 (64 from x, 256 from h).
    for (int t = 0; t < 168; ++t) {
        lstm_step<10, true><<<dim3(32, 16), 128, 0, stream>>>(
            x + (size_t)t * BATCH * 64, Hc, Hn, C, WfE, bE);
        _Float16* tmp = Hc; Hc = Hn; Hn = tmp;
    }

    // Decoder: 24 steps, input == h so K = 256 with summed weights.
    for (int s = 0; s < 24; ++s) {
        lstm_step<8, false><<<dim3(32, 16), 128, 0, stream>>>(
            nullptr, Hc, Hn, C, WfD, bD);
        linear_out<<<8, 256, 0, stream>>>(Hn, W_lin, b_lin, out + (size_t)s * BATCH);
        _Float16* tmp = Hc; Hc = Hn; Hn = tmp;
    }
}
